// GraphModule_73443940762192
// MI455X (gfx1250) — matrix operand order repaired and checked by reading, not compiled, not run on hardware
//
#include <hip/hip_runtime.h>
#include <cstdint>
#include <cstddef>

typedef unsigned short u16;
typedef __attribute__((ext_vector_type(16))) __bf16 v16bf;
typedef __attribute__((ext_vector_type(8)))  float  v8f;
typedef __attribute__((ext_vector_type(4)))  unsigned u32x4;
typedef __attribute__((ext_vector_type(8)))  unsigned u32x8;

constexpr int BATCH = 4, SEQ = 1536, DIM = 768, HEADS = 24, HDIM = 32, FDIM = 3072;
constexpr int MTOT  = BATCH * SEQ;                  // 6144 rows
constexpr float ATT_SCALE = 0.17677669529663687f;   // 1/sqrt(32)

__device__ __forceinline__ u16 f32_to_bf16(float f) {
  unsigned u = __float_as_uint(f);
  u += 0x7fffu + ((u >> 16) & 1u);                  // round-to-nearest-even
  return (u16)(u >> 16);
}

__device__ __forceinline__ unsigned pack_bf16x2(float lo, float hi) {
  return (unsigned)f32_to_bf16(lo) | ((unsigned)f32_to_bf16(hi) << 16);
}

union V16U { v16bf v; uint4 q[2]; };

// Exact CDNA5 16-bit A-matrix fragment layout (16x32, MxK):
// lanes 0-15 hold M=lane, K chunks [0..7] and [16..23];
// lanes 16-31 hold M=lane-16, K chunks [8..15] and [24..31].
__device__ __forceinline__ v16bf load_a_frag(const u16* rowbase, int lane) {
  const u16* b = rowbase + ((lane >> 4) << 3);
  V16U t;
  t.q[0] = *(const uint4*)(b);
  t.q[1] = *(const uint4*)(b + 16);
  return t.v;
}

__device__ __forceinline__ v8f wmma_bf16(v16bf a, v16bf b, v8f c) {
  return __builtin_amdgcn_wmma_f32_16x16x32_bf16(false, a, false, b, (short)0, c,
                                                 false, false);
}

// 2x async global->LDS b128 copies (ASYNCcnt path); lA is an LDS byte offset.
#define ASYNC_CP2(lA, gA)                                                      \
  asm volatile("global_load_async_to_lds_b128 %0, %2, off\n\t"                 \
               "global_load_async_to_lds_b128 %1, %3, off"                     \
               :: "v"(lA), "v"((unsigned)((lA) + 16)),                          \
                  "v"(gA), "v"((gA) + 8)                                        \
               : "memory")

#define WAIT_ASYNC0() asm volatile("s_wait_asynccnt 0" ::: "memory")

// ---------------------------------------------------------------------------
// Tensor Data Mover: 2D tile (tile_w x tile_h elements of 2 bytes) from a
// row-major tensor with row stride `stride_w` elements into LDS at lds_off.
// Builds the D# (group0 + group1) in SGPR tuples and issues the 2-group form
// of tensor_load_to_lds (VADDR2/3 = NULL). Tracked by TENSORcnt.
// ---------------------------------------------------------------------------
__device__ __forceinline__ void tdm_load_2d(unsigned lds_off, const void* gtile,
                                            unsigned tile_w, unsigned tile_h,
                                            unsigned tensor_w, unsigned tensor_h,
                                            unsigned stride_w) {
  unsigned long long ga = (unsigned long long)(size_t)gtile;
  u32x4 g0;
  g0[0] = 1u;                                            // count=1, user D#
  g0[1] = lds_off;                                       // lds_addr (bytes)
  g0[2] = (unsigned)ga;                                  // global_addr[31:0]
  g0[3] = (unsigned)((ga >> 32) & 0x01ffffffu)           // global_addr[56:32]
        | (2u << 30);                                    // type = 2 ("image")
  u32x8 g1;
  g1[0] = 1u << 16;                                      // wg_mask=0, data_size=2B
  g1[1] = (tensor_w & 0xffffu) << 16;                    // tensor_dim0 lo
  g1[2] = (tensor_w >> 16) | ((tensor_h & 0xffffu) << 16);
  g1[3] = (tensor_h >> 16) | ((tile_w & 0xffffu) << 16); // tensor_dim1 hi | tile_dim0
  g1[4] = tile_h & 0xffffu;                              // tile_dim1, tile_dim2=0
  g1[5] = stride_w;                                      // tensor_dim0_stride lo
  g1[6] = 0u;                                            // stride hi / dim1_stride
  g1[7] = 0u;
  asm volatile("tensor_load_to_lds %0, %1" :: "s"(g0), "s"(g1) : "memory");
}

// ---------------------------------------------------------------------------
// Embedding gather: x[m,:] = wte[tok[m],:] + wpe[pos[m],:]
// ---------------------------------------------------------------------------
__global__ __launch_bounds__(256) void embed_k(const int* __restrict__ tok,
                                               const int* __restrict__ pos,
                                               const float* __restrict__ wte,
                                               const float* __restrict__ wpe,
                                               float* __restrict__ x) {
  const int m = blockIdx.x;
  const float* wt = wte + (size_t)tok[m] * DIM;
  const float* wp = wpe + (size_t)pos[m] * DIM;
  float* xr = x + (size_t)m * DIM;
  for (int d = threadIdx.x; d < DIM; d += 256) xr[d] = wt[d] + wp[d];
}

// ---------------------------------------------------------------------------
// LayerNorm over D=768 per row; outputs bf16 (GEMM input) or f32 (final).
// ---------------------------------------------------------------------------
__global__ __launch_bounds__(256) void layernorm_k(const float* __restrict__ x,
                                                   const float* __restrict__ g,
                                                   const float* __restrict__ bt,
                                                   float* __restrict__ outf,
                                                   u16* __restrict__ outb) {
  __shared__ float rs[8], rq[8];
  const int m = blockIdx.x, tid = threadIdx.x, lane = tid & 31, wave = tid >> 5;
  const float* xr = x + (size_t)m * DIM;
  float s = 0.f, s2 = 0.f;
  for (int d = tid; d < DIM; d += 256) { float v = xr[d]; s += v; s2 += v * v; }
#pragma unroll
  for (int off = 16; off; off >>= 1) {
    s  += __shfl_xor(s,  off, 32);
    s2 += __shfl_xor(s2, off, 32);
  }
  if (lane == 0) { rs[wave] = s; rq[wave] = s2; }
  __syncthreads();
  if (tid == 0) {
    float a = 0.f, c = 0.f;
    for (int i = 0; i < 8; ++i) { a += rs[i]; c += rq[i]; }
    float mu = a / DIM;
    rs[0] = mu;
    rq[0] = fmaxf(c / DIM - mu * mu, 0.f);
  }
  __syncthreads();
  const float mu = rs[0], rinv = rsqrtf(rq[0] + 1e-5f);
  for (int d = tid; d < DIM; d += 256) {
    float v = (xr[d] - mu) * rinv * g[d] + bt[d];
    if (outb) outb[(size_t)m * DIM + d] = f32_to_bf16(v);
    else      outf[(size_t)m * DIM + d] = v;
  }
}

// ---------------------------------------------------------------------------
// f32 -> bf16 weight conversion
// ---------------------------------------------------------------------------
__global__ __launch_bounds__(256) void cvt_bf16_k(const float* __restrict__ w,
                                                  u16* __restrict__ o, int n) {
  int i = blockIdx.x * 256 + threadIdx.x;
  if (i < n) o[i] = f32_to_bf16(w[i]);
}

// ---------------------------------------------------------------------------
// Tiled WMMA GEMM: C[M,N] = A[M,K] @ W[N,K]^T + bias, fused epilogues.
//   EPI 0: scatter to q (scaled), k, vT (bf16)  [QKV projection]
//   EPI 1: + residual -> f32                    [attn proj / FC2]
//   EPI 2: GELU -> bf16                         [FC1]
// 256 threads = 8 waves; macro tile 128x128; BK=32; wave tile 32x64 = 8 wmma.
// Double-buffered staging: A tile via per-lane async-to-LDS (ASYNCcnt),
// B (weights) tile via Tensor Data Mover descriptor DMA (TENSORcnt, wave 0).
// ---------------------------------------------------------------------------
template <int EPI>
__global__ __launch_bounds__(256) void gemm_bf16_k(
    const u16* __restrict__ A, const u16* __restrict__ W,
    const float* __restrict__ bias, const float* __restrict__ R,
    void* __restrict__ o0, void* __restrict__ o1, void* __restrict__ o2,
    int M, int N, int K) {
  __shared__ u16 As[2][128][32];
  __shared__ u16 Bs[2][128][32];
  const int tid = threadIdx.x, lane = tid & 31, wave = tid >> 5;
  const int bm = blockIdx.y * 128, bn = blockIdx.x * 128;
  const int wr = (wave >> 1) * 32;  // 4 row groups of 32
  const int wc = (wave & 1) * 64;   // 2 col groups of 64
  v8f acc[2][4] = {};

  // A staging: thread owns 16 halfs (32B) of one row
  const int sr = tid >> 1, sc = (tid & 1) << 4;
  const u16* gA = A + (size_t)(bm + sr) * K + sc;
  const u16* gBtile = W + (size_t)bn * K;  // corner of this block's weight rows
  unsigned ldsA[2] = { (unsigned)(size_t)&As[0][sr][sc],
                       (unsigned)(size_t)&As[1][sr][sc] };
  unsigned ldsB[2] = { (unsigned)(size_t)&Bs[0][0][0],
                       (unsigned)(size_t)&Bs[1][0][0] };
  const bool tdmw = (wave == 0);  // wave-uniform: wave 0 drives the TDM

  ASYNC_CP2(ldsA[0], gA);
  if (tdmw)
    tdm_load_2d(ldsB[0], gBtile, 32, 128, (unsigned)K, 1u << 20, (unsigned)K);

  for (int k0 = 0; k0 < K; k0 += 32) {
    const int buf = (k0 >> 5) & 1;
    if (tdmw) __builtin_amdgcn_s_wait_tensorcnt(0);  // B tile landed
    WAIT_ASYNC0();                                   // own A copies landed
    __syncthreads();                                 // stage `buf` ready for all
    if (k0 + 32 < K) {                               // prefetch next slab
      ASYNC_CP2(ldsA[buf ^ 1], gA + k0 + 32);
      if (tdmw)
        tdm_load_2d(ldsB[buf ^ 1], gBtile + k0 + 32, 32, 128, (unsigned)K,
                    1u << 20, (unsigned)K);
    }

    v16bf a0 = load_a_frag(&As[buf][wr + (lane & 15)][0], lane);
    v16bf a1 = load_a_frag(&As[buf][wr + 16 + (lane & 15)][0], lane);
    v16bf bfr[4];
#pragma unroll
    for (int cj = 0; cj < 4; ++cj)
      bfr[cj] = load_a_frag(&Bs[buf][wc + cj * 16 + (lane & 15)][0], lane);
#pragma unroll
    for (int cj = 0; cj < 4; ++cj) {
      acc[0][cj] = wmma_bf16(a0, bfr[cj], acc[0][cj]);
      acc[1][cj] = wmma_bf16(a1, bfr[cj], acc[1][cj]);
    }
  }

  // C layout: VGPR i -> M = i + 8*(lane>>4), N = lane&15 (per 16x16 tile)
#pragma unroll
  for (int ri = 0; ri < 2; ++ri)
#pragma unroll
    for (int cj = 0; cj < 4; ++cj)
#pragma unroll
      for (int i = 0; i < 8; ++i) {
        int m = bm + wr + ri * 16 + i + ((lane >> 4) << 3);
        int n = bn + wc + cj * 16 + (lane & 15);
        float v = acc[ri][cj][i] + bias[n];
        if constexpr (EPI == 0) {
          int b = m / SEQ, s = m - b * SEQ;
          int head = n / (3 * HDIM), r = n - head * (3 * HDIM);
          if (r < HDIM) {  // Q, pre-scaled by 1/sqrt(DH)
            ((u16*)o0)[(((size_t)b * HEADS + head) * SEQ + s) * HDIM + r] =
                f32_to_bf16(v * ATT_SCALE);
          } else if (r < 2 * HDIM) {  // K
            ((u16*)o1)[(((size_t)b * HEADS + head) * SEQ + s) * HDIM + (r - HDIM)] =
                f32_to_bf16(v);
          } else {  // V transposed -> [B,H,DH,S]
            ((u16*)o2)[(((size_t)b * HEADS + head) * HDIM + (r - 2 * HDIM)) * SEQ + s] =
                f32_to_bf16(v);
          }
        } else if constexpr (EPI == 1) {
          ((float*)o0)[(size_t)m * N + n] = v + R[(size_t)m * N + n];
        } else {
          float gl = 0.5f * v * (1.0f + erff(v * 0.70710678118654752f));
          ((u16*)o0)[(size_t)m * N + n] = f32_to_bf16(gl);
        }
      }
}

// ---------------------------------------------------------------------------
// Flash attention, transposed-score form. One wave per (b, head, 16-q tile).
// Per 32-key chunk:  S^T = K_tile x Q^T  (2 wmma)  -> each lane owns ONE query
// column, 16 of its key scores in-register => softmax stats are in-lane VALU
// plus a single shfl_xor(16) each; then ctx^T += V^T x P (2 wmma) with P
// restaged through LDS as a B-fragment (two packed b128 stores).
// ---------------------------------------------------------------------------
__global__ __launch_bounds__(256) void flash_attn_k(
    const u16* __restrict__ q, const u16* __restrict__ k,
    const u16* __restrict__ vT, const unsigned char* __restrict__ mask,
    u16* __restrict__ ctx) {
  __shared__ u16 Ps[8][16][32];  // per-wave P staging (8 KB total)
  const int lane = threadIdx.x & 31, wave = threadIdx.x >> 5;
  const int task = blockIdx.x * 8 + wave;
  const int qt = task % (SEQ / 16);
  const int hh = (task / (SEQ / 16)) % HEADS;
  const int b  = task / ((SEQ / 16) * HEADS);
  const int qbase = qt * 16;
  const int qcol  = lane & 15;          // query column owned by this lane
  const int khalf = (lane >> 4) << 3;   // key sub-block base (0 or 8)
  const u16* qp = q  + (((size_t)b * HEADS + hh) * SEQ + qbase) * HDIM;
  const u16* kp = k  + (((size_t)b * HEADS + hh) * SEQ) * HDIM;
  const u16* vp = vT + (((size_t)b * HEADS + hh) * HDIM) * SEQ;
  const unsigned char* mrow =
      mask + (size_t)b * SEQ * SEQ + (size_t)(qbase + qcol) * SEQ;

  // Q as B-fragment (constant across all key chunks)
  v16bf qfb = load_a_frag(&qp[(size_t)qcol * HDIM], lane);

  v8f o0 = {}, o1 = {};          // ctx^T accumulators: dh 0..15 / 16..31
  float rm = -3.0e38f, rl = 0.f; // per-query running max / sum (scalar!)

  for (int kt = 0; kt < SEQ; kt += 32) {
    v16bf ka0 = load_a_frag(&kp[(size_t)(kt + qcol) * HDIM], lane);
    v16bf ka1 = load_a_frag(&kp[(size_t)(kt + 16 + qcol) * HDIM], lane);
    v8f z = {};
    v8f st0 = wmma_bf16(ka0, qfb, z);  // S^T: M = key kt..kt+15, N = query
    v8f st1 = wmma_bf16(ka1, qfb, z);  // S^T: M = key kt+16..kt+31

    // mask: two contiguous 8-byte runs of this query's mask row
    unsigned long long mw0 = *(const unsigned long long*)(mrow + kt + khalf);
    unsigned long long mw1 = *(const unsigned long long*)(mrow + kt + 16 + khalf);

    float e0[8], e1[8];
    float vmax = -3.0e38f;
#pragma unroll
    for (int i = 0; i < 8; ++i) {
      float a = st0[i], c = st1[i];
      if ((mw0 >> (8 * i)) & 0xffull) a = -10000.f;
      if ((mw1 >> (8 * i)) & 0xffull) c = -10000.f;
      e0[i] = a; e1[i] = c;
      vmax = fmaxf(vmax, fmaxf(a, c));
    }
    vmax = fmaxf(vmax, __shfl_xor(vmax, 16, 32));  // combine key halves
    float mnew = fmaxf(rm, vmax);
    float scl  = __expf(rm - mnew);
    float p0[8], p1[8], ps = 0.f;
#pragma unroll
    for (int i = 0; i < 8; ++i) {
      p0[i] = __expf(e0[i] - mnew);
      p1[i] = __expf(e1[i] - mnew);
      ps += p0[i] + p1[i];
    }
    ps += __shfl_xor(ps, 16, 32);
    rl = rl * scl + ps;
    rm = mnew;
#pragma unroll
    for (int i = 0; i < 8; ++i) { o0[i] *= scl; o1[i] *= scl; }

    // stage P[q][key] (two packed b128 LDS stores; same-wave DS is in-order)
    uint4 w;
    w.x = pack_bf16x2(p0[0], p0[1]); w.y = pack_bf16x2(p0[2], p0[3]);
    w.z = pack_bf16x2(p0[4], p0[5]); w.w = pack_bf16x2(p0[6], p0[7]);
    *(uint4*)&Ps[wave][qcol][khalf] = w;
    w.x = pack_bf16x2(p1[0], p1[1]); w.y = pack_bf16x2(p1[2], p1[3]);
    w.z = pack_bf16x2(p1[4], p1[5]); w.w = pack_bf16x2(p1[6], p1[7]);
    *(uint4*)&Ps[wave][qcol][16 + khalf] = w;

    v16bf pb  = load_a_frag(&Ps[wave][qcol][0], lane);
    v16bf va0 = load_a_frag(&vp[(size_t)qcol * SEQ + kt], lane);
    v16bf va1 = load_a_frag(&vp[(size_t)(qcol + 16) * SEQ + kt], lane);
    o0 = wmma_bf16(va0, pb, o0);  // ctx^T rows dh 0..15
    o1 = wmma_bf16(va1, pb, o1);  // ctx^T rows dh 16..31
  }

  const float inv = 1.0f / rl;
  const size_t base = ((size_t)b * SEQ + qbase + qcol) * DIM + hh * HDIM;
  uint4 w;
  w.x = pack_bf16x2(o0[0] * inv, o0[1] * inv);
  w.y = pack_bf16x2(o0[2] * inv, o0[3] * inv);
  w.z = pack_bf16x2(o0[4] * inv, o0[5] * inv);
  w.w = pack_bf16x2(o0[6] * inv, o0[7] * inv);
  *(uint4*)&ctx[base + khalf] = w;
  w.x = pack_bf16x2(o1[0] * inv, o1[1] * inv);
  w.y = pack_bf16x2(o1[2] * inv, o1[3] * inv);
  w.z = pack_bf16x2(o1[4] * inv, o1[5] * inv);
  w.w = pack_bf16x2(o1[6] * inv, o1[7] * inv);
  *(uint4*)&ctx[base + 16 + khalf] = w;
}

// ---------------------------------------------------------------------------
extern "C" void kernel_launch(void* const* d_in, const int* in_sizes, int n_in,
                              void* d_out, int out_size, void* d_ws, size_t ws_size,
                              hipStream_t stream) {
  (void)in_sizes; (void)n_in; (void)out_size; (void)ws_size;
  const int* tok   = (const int*)d_in[0];
  const int* pos   = (const int*)d_in[1];
  const unsigned char* mask = (const unsigned char*)d_in[2];
  const float* wte  = (const float*)d_in[3];
  const float* wpe  = (const float*)d_in[4];
  const float* ln1w = (const float*)d_in[5];
  const float* ln1b = (const float*)d_in[6];
  const float* qkvw = (const float*)d_in[7];
  const float* qkvb = (const float*)d_in[8];
  const float* prjw = (const float*)d_in[9];
  const float* prjb = (const float*)d_in[10];
  const float* ln2w = (const float*)d_in[11];
  const float* ln2b = (const float*)d_in[12];
  const float* fc1w = (const float*)d_in[13];
  const float* fc1b = (const float*)d_in[14];
  const float* fc2w = (const float*)d_in[15];
  const float* fc2b = (const float*)d_in[16];
  const float* ln3w = (const float*)d_in[17];
  const float* ln3b = (const float*)d_in[18];

  char* p = (char*)d_ws;
  auto carve = [&](size_t bytes) -> void* {
    void* r = (void*)p;
    p += (bytes + 255) & ~(size_t)255;
    return r;
  };
  float* x   = (float*)carve((size_t)MTOT * DIM * 4);
  float* x2  = (float*)carve((size_t)MTOT * DIM * 4);
  float* x3  = (float*)carve((size_t)MTOT * DIM * 4);
  u16*   h1  = (u16*)carve((size_t)MTOT * DIM * 2);
  u16*   h2  = (u16*)carve((size_t)MTOT * DIM * 2);
  u16*   ctx = (u16*)carve((size_t)MTOT * DIM * 2);
  u16*   act = (u16*)carve((size_t)MTOT * FDIM * 2);
  u16*   qb  = (u16*)carve((size_t)BATCH * HEADS * SEQ * HDIM * 2);
  u16*   kb  = (u16*)carve((size_t)BATCH * HEADS * SEQ * HDIM * 2);
  u16*   vTb = (u16*)carve((size_t)BATCH * HEADS * SEQ * HDIM * 2);
  u16*   wq  = (u16*)carve((size_t)3 * DIM * DIM * 2);
  u16*   wp  = (u16*)carve((size_t)DIM * DIM * 2);
  u16*   w1  = (u16*)carve((size_t)FDIM * DIM * 2);
  u16*   w2  = (u16*)carve((size_t)DIM * FDIM * 2);

  // 1) embeddings + first LN (bf16 activations for the matrix pipe)
  embed_k<<<MTOT, 256, 0, stream>>>(tok, pos, wte, wpe, x);
  layernorm_k<<<MTOT, 256, 0, stream>>>(x, ln1w, ln1b, nullptr, h1);

  // 2) weights -> bf16
  int nq = 3 * DIM * DIM, np = DIM * DIM, n1 = FDIM * DIM, n2 = DIM * FDIM;
  cvt_bf16_k<<<(nq + 255) / 256, 256, 0, stream>>>(qkvw, wq, nq);
  cvt_bf16_k<<<(np + 255) / 256, 256, 0, stream>>>(prjw, wp, np);
  cvt_bf16_k<<<(n1 + 255) / 256, 256, 0, stream>>>(fc1w, w1, n1);
  cvt_bf16_k<<<(n2 + 255) / 256, 256, 0, stream>>>(fc2w, w2, n2);

  // 3) QKV projection with q/k/vT scatter epilogue
  gemm_bf16_k<0><<<dim3((3 * DIM) / 128, MTOT / 128), 256, 0, stream>>>(
      h1, wq, qkvb, nullptr, qb, kb, vTb, MTOT, 3 * DIM, DIM);

  // 4) flash attention (9216 wave-tasks, 8 per block)
  flash_attn_k<<<(BATCH * HEADS * (SEQ / 16)) / 8, 256, 0, stream>>>(
      qb, kb, vTb, mask, ctx);

  // 5) output projection + residual
  gemm_bf16_k<1><<<dim3(DIM / 128, MTOT / 128), 256, 0, stream>>>(
      ctx, wp, prjb, x, x2, nullptr, nullptr, MTOT, DIM, DIM);

  // 6) MLP
  layernorm_k<<<MTOT, 256, 0, stream>>>(x2, ln2w, ln2b, nullptr, h2);
  gemm_bf16_k<2><<<dim3(FDIM / 128, MTOT / 128), 256, 0, stream>>>(
      h2, w1, fc1b, nullptr, act, nullptr, nullptr, MTOT, FDIM, DIM);
  gemm_bf16_k<1><<<dim3(DIM / 128, MTOT / 128), 256, 0, stream>>>(
      act, w2, fc2b, x2, x3, nullptr, nullptr, MTOT, DIM, FDIM);

  // 7) final LN -> f32 output
  layernorm_k<<<MTOT, 256, 0, stream>>>(x3, ln3w, ln3b, (float*)d_out, nullptr);
}
